// MoEBlock_13537736917776
// MI455X (gfx1250) — compile-verified, matrix operand order checked by reading, not run on hardware
//
#include <hip/hip_runtime.h>
#include <hip/hip_bf16.h>

// ---------------------------------------------------------------------------
// MoE block for MI455X (gfx1250, wave32, WMMA).
// Sparse dispatch (top-2 of 8 experts + shared expert as "expert 8"),
// f16 weights/activations with f32 WMMA accumulation.
// Data path: async global->LDS (double buffered, ASYNCcnt) when the
// toolchain exposes the gfx1250 async builtins; classic load/ds_store
// fallback otherwise.
// ---------------------------------------------------------------------------

#define DIMD   1024
#define INNERD 4096
#define NEXP   8
#define TOK    4096   // B*S
#define CAP    4096   // max tokens per expert list

#define BM 64
#define BN 128
#define BK 32
#define KP 40         // padded LDS row: 32 halves + 8 pad = 80B (16B aligned, bank-conflict free)

typedef _Float16 v16h __attribute__((ext_vector_type(16)));
typedef float    v8f  __attribute__((ext_vector_type(8)));
typedef int      v4i  __attribute__((ext_vector_type(4)));

union FragH { v16h v; uint4 q[2]; };

#if defined(__has_builtin)
#  if __has_builtin(__builtin_amdgcn_global_load_async_to_lds_b128) && \
      __has_builtin(__builtin_amdgcn_s_wait_asynccnt)
#    define HAS_ASYNC_LDS 1
#  endif
#endif
#ifndef HAS_ASYNC_LDS
#  define HAS_ASYNC_LDS 0
#endif

#if HAS_ASYNC_LDS
__device__ __forceinline__ void async_cp16(const _Float16* g, _Float16* l) {
    // one wave-level instruction: 32 lanes x 16B, tracked with ASYNCcnt
    __builtin_amdgcn_global_load_async_to_lds_b128(
        (__attribute__((address_space(1))) v4i*)(g),
        (__attribute__((address_space(3))) v4i*)(l), 0, 0);
}
#endif

__device__ __forceinline__ float gelu_exact(float x) {
    return 0.5f * x * (1.0f + erff(x * 0.70710678118654752f));
}

// ---------------------------------------------------------------------------
// 0) zero per-expert counters
// ---------------------------------------------------------------------------
__global__ void zero_counts_kernel(int* __restrict__ counts) {
    if (threadIdx.x < 9) counts[threadIdx.x] = 0;
}

// ---------------------------------------------------------------------------
// 1) gate: logits -> softmax -> top2 -> slot assignment (slot order is
//    arbitrary but tracked per token, so final values are permutation-
//    invariant => deterministic, no float atomics anywhere)
// ---------------------------------------------------------------------------
__global__ __launch_bounds__(256) void gate_route_kernel(
        const float* __restrict__ x, const float* __restrict__ wg,
        int* __restrict__ counts, int* __restrict__ idxList,
        int4* __restrict__ route, float2* __restrict__ routeW) {
    const int t   = blockIdx.x;
    const int tid = threadIdx.x;
    __shared__ __align__(16) float xrow[DIMD];
    __shared__ float lg[NEXP];

    ((float4*)xrow)[tid] = ((const float4*)(x + (size_t)t * DIMD))[tid];
    __syncthreads();

    const int wave = tid >> 5, lane = tid & 31;
    const float* wrow = wg + wave * DIMD;
    float s = 0.0f;
    for (int k = lane; k < DIMD; k += 32) s += xrow[k] * wrow[k];
    for (int off = 16; off >= 1; off >>= 1) s += __shfl_xor(s, off, 32);
    if (lane == 0) lg[wave] = s;
    __syncthreads();

    if (tid == 0) {
        float m = lg[0];
        for (int e = 1; e < NEXP; ++e) m = fmaxf(m, lg[e]);
        float p[NEXP]; float sum = 0.0f;
        for (int e = 0; e < NEXP; ++e) { p[e] = expf(lg[e] - m); sum += p[e]; }
        const float inv = 1.0f / sum;
        int e0 = 0; float p0 = p[0];
        for (int e = 1; e < NEXP; ++e) if (p[e] > p0) { p0 = p[e]; e0 = e; }
        int e1 = -1; float p1 = -1.0f;
        for (int e = 0; e < NEXP; ++e) if (e != e0 && p[e] > p1) { p1 = p[e]; e1 = e; }
        p0 *= inv; p1 *= inv;
        int pos0 = atomicAdd(&counts[e0], 1); idxList[e0 * CAP + pos0] = t;
        int pos1 = atomicAdd(&counts[e1], 1); idxList[e1 * CAP + pos1] = t;
        route[t]  = make_int4(e0, pos0, e1, pos1);
        routeW[t] = make_float2(p0, p1);
    }
}

// ---------------------------------------------------------------------------
// 2) prefix offsets: offs[0..8], offs[8] = total routed slots (2*TOK)
// ---------------------------------------------------------------------------
__global__ void offsets_kernel(const int* __restrict__ counts, int* __restrict__ offs) {
    if (threadIdx.x == 0) {
        offs[0] = 0;
        for (int e = 0; e < NEXP; ++e) offs[e + 1] = offs[e] + counts[e];
    }
}

// ---------------------------------------------------------------------------
// 3) f32 -> f16 convert (vectorized)
// ---------------------------------------------------------------------------
__global__ void cvt_f32_f16_kernel(const float* __restrict__ in, _Float16* __restrict__ out, int n4) {
    int i = blockIdx.x * blockDim.x + threadIdx.x;
    if (i < n4) {
        float4 v = ((const float4*)in)[i];
        union { _Float16 h[4]; uint2 u; } p;
        p.h[0] = (_Float16)v.x; p.h[1] = (_Float16)v.y;
        p.h[2] = (_Float16)v.z; p.h[3] = (_Float16)v.w;
        ((uint2*)out)[i] = p.u;
    }
}

// ---------------------------------------------------------------------------
// 4) batched transpose + convert: in [b][R][C] f32 -> out [b][C][R] f16
//    (weights become [N][K] so B fragments are contiguous ds_read_b128s)
// ---------------------------------------------------------------------------
__global__ __launch_bounds__(256) void transpose_cvt_kernel(
        const float* __restrict__ in, _Float16* __restrict__ out, int R, int C) {
    __shared__ float tile[32][33];
    const int b = blockIdx.z;
    const float* src = in + (size_t)b * R * C;
    _Float16*    dst = out + (size_t)b * R * C;
    const int c0 = blockIdx.x * 32, r0 = blockIdx.y * 32;
    const int tx = threadIdx.x, ty = threadIdx.y;
    #pragma unroll
    for (int i = 0; i < 32; i += 8)
        tile[ty + i][tx] = src[(size_t)(r0 + ty + i) * C + (c0 + tx)];
    __syncthreads();
    #pragma unroll
    for (int i = 0; i < 32; i += 8)
        dst[(size_t)(c0 + ty + i) * R + (r0 + tx)] = (_Float16)tile[tx][ty + i];
}

// ---------------------------------------------------------------------------
// 5/6) unified FFN GEMM.  STAGE 1: H = gelu(gather(X) @ W1 + b1) -> f16
//      STAGE 2: Y = H @ W2 + b2 -> f32 (contiguous slot rows)
//      64x128 block tile, 8 waves (4x2), 1 v_wmma_f32_16x16x32_f16 per
//      16x16 tile per K-step; A fragment reused across the 4 N-tiles.
// ---------------------------------------------------------------------------
template <int STAGE>
__global__ __launch_bounds__(256) void ffn_gemm_kernel(
        const _Float16* __restrict__ Aglob,   // Xh (stage1) / Hbuf (stage2)
        const _Float16* __restrict__ Wexp,    // W1T / W2T   ([e][N][K] f16)
        const _Float16* __restrict__ Wsh,     // sW1T / sW2T
        const float* __restrict__ bexp, const float* __restrict__ bsh,
        const int* __restrict__ counts, const int* __restrict__ offs,
        const int* __restrict__ idxList,
        _Float16* __restrict__ Hout, float* __restrict__ Yout) {
    constexpr int KD   = (STAGE == 1) ? DIMD : INNERD;
    constexpr int ND   = (STAGE == 1) ? INNERD : DIMD;
    constexpr int ABUF = BM * KP;
    constexpr int BBUF = BN * KP;
    constexpr int NT   = KD / BK;

    const int e     = blockIdx.z;
    const int mBase = blockIdx.y * BM;
    const int nBase = blockIdx.x * BN;
    const int cnt   = (e < NEXP) ? counts[e] : TOK;
    if (mBase >= cnt) return;                       // uniform -> full EXEC kept
    const int slotBase = offs[e];
    const _Float16* Bmat = (e < NEXP) ? (Wexp + (size_t)e * DIMD * INNERD) : Wsh;
    const float*    bias = (e < NEXP) ? (bexp + (size_t)e * ND) : bsh;

    __shared__ __align__(16) _Float16 At[2 * ABUF];
    __shared__ __align__(16) _Float16 Bt[2 * BBUF];
    __shared__ int rowIdx[BM];

    const int tid = threadIdx.x;
    if constexpr (STAGE == 1) {
        if (tid < BM) {
            const int slot = mBase + tid;
            int r = 0;
            if (e == NEXP) r = slot;                          // shared expert: identity
            else if (slot < cnt) r = idxList[e * CAP + slot]; // gathered token
            rowIdx[tid] = r;
        }
        __syncthreads();
    }

    const int lane = tid & 31;
    const int wave = tid >> 5;
    const int wm   = wave & 3;    // 4 x 16-row groups
    const int wn   = wave >> 2;   // 2 x 64-col groups
    const int arow = tid >> 2, aseg = tid & 3;

    const _Float16* aSrc;
    if constexpr (STAGE == 1) aSrc = Aglob + (size_t)rowIdx[arow] * KD + aseg * 8;
    else                      aSrc = Aglob + (size_t)(slotBase + mBase + arow) * KD + aseg * 8;
    const _Float16* b0Src = Bmat + (size_t)(nBase + arow) * KD + aseg * 8;
    const _Float16* b1Src = Bmat + (size_t)(nBase + arow + 64) * KD + aseg * 8;

    v8f acc[4];
    const v8f vz = {0.f,0.f,0.f,0.f,0.f,0.f,0.f,0.f};
    #pragma unroll
    for (int t = 0; t < 4; ++t) acc[t] = vz;

    const int m16 = lane & 15, kh = (lane >> 4) * 8;
    const int mr  = wm * 16 + m16;
    const int nr0 = wn * 64 + m16;

    // load all fragments first, then issue the 4 WMMAs back-to-back
    auto compute_tile = [&](int cur) {
        const _Float16* Ab = At + cur * ABUF;
        const _Float16* Bb = Bt + cur * BBUF;
        FragH aF, bF0, bF1, bF2, bF3;
        aF.q[0]  = *(const uint4*)(Ab + mr * KP + kh);
        aF.q[1]  = *(const uint4*)(Ab + mr * KP + 16 + kh);
        bF0.q[0] = *(const uint4*)(Bb + (nr0     ) * KP + kh);
        bF0.q[1] = *(const uint4*)(Bb + (nr0     ) * KP + 16 + kh);
        bF1.q[0] = *(const uint4*)(Bb + (nr0 + 16) * KP + kh);
        bF1.q[1] = *(const uint4*)(Bb + (nr0 + 16) * KP + 16 + kh);
        bF2.q[0] = *(const uint4*)(Bb + (nr0 + 32) * KP + kh);
        bF2.q[1] = *(const uint4*)(Bb + (nr0 + 32) * KP + 16 + kh);
        bF3.q[0] = *(const uint4*)(Bb + (nr0 + 48) * KP + kh);
        bF3.q[1] = *(const uint4*)(Bb + (nr0 + 48) * KP + 16 + kh);
        acc[0] = __builtin_amdgcn_wmma_f32_16x16x32_f16(false, aF.v, false, bF0.v, (short)0, acc[0], false, false);
        acc[1] = __builtin_amdgcn_wmma_f32_16x16x32_f16(false, aF.v, false, bF1.v, (short)0, acc[1], false, false);
        acc[2] = __builtin_amdgcn_wmma_f32_16x16x32_f16(false, aF.v, false, bF2.v, (short)0, acc[2], false, false);
        acc[3] = __builtin_amdgcn_wmma_f32_16x16x32_f16(false, aF.v, false, bF3.v, (short)0, acc[3], false, false);
    };

#if HAS_ASYNC_LDS
    // -------- async global->LDS, double-buffered, tracked with ASYNCcnt ----
    async_cp16(aSrc,  At + arow * KP + aseg * 8);
    async_cp16(b0Src, Bt + arow * KP + aseg * 8);
    async_cp16(b1Src, Bt + (arow + 64) * KP + aseg * 8);
    #pragma unroll 2
    for (int it = 0; it < NT; ++it) {
        const int cur = it & 1, nxt = cur ^ 1;
        if (it + 1 < NT) {
            const int k0n = (it + 1) * BK;
            async_cp16(aSrc + k0n,  At + nxt * ABUF + arow * KP + aseg * 8);
            async_cp16(b0Src + k0n, Bt + nxt * BBUF + arow * KP + aseg * 8);
            async_cp16(b1Src + k0n, Bt + nxt * BBUF + (arow + 64) * KP + aseg * 8);
            __builtin_amdgcn_s_wait_asynccnt(3);   // current tile done (in-order), next in flight
        } else {
            __builtin_amdgcn_s_wait_asynccnt(0);
        }
        __syncthreads();
        compute_tile(cur);
        __syncthreads();                            // before buf `cur` is re-issued next iter
    }
#else
    // -------- fallback: global->VGPR->ds_store -------
    for (int k0 = 0; k0 < KD; k0 += BK) {
        uint4 aV  = *(const uint4*)(aSrc + k0);
        uint4 bV0 = *(const uint4*)(b0Src + k0);
        uint4 bV1 = *(const uint4*)(b1Src + k0);
        if (k0 + BK < KD) {
            __builtin_prefetch(b0Src + k0 + BK, 0, 1);
            __builtin_prefetch(b1Src + k0 + BK, 0, 1);
        }
        __syncthreads();
        *(uint4*)(At + arow * KP + aseg * 8)        = aV;
        *(uint4*)(Bt + arow * KP + aseg * 8)        = bV0;
        *(uint4*)(Bt + (arow + 64) * KP + aseg * 8) = bV1;
        __syncthreads();
        compute_tile(0);
    }
#endif

    // epilogue (C layout: n = lane%16, m = 8*(lane/16)+j)
    const int mloc  = wm * 16 + (lane >> 4) * 8;
    const int ncol0 = nBase + wn * 64 + m16;
    #pragma unroll
    for (int t = 0; t < 4; ++t) {
        const int n  = ncol0 + t * 16;
        const float bv = bias[n];
        #pragma unroll
        for (int j = 0; j < 8; ++j) {
            const int m = mBase + mloc + j;
            if (m < cnt) {
                if constexpr (STAGE == 1) {
                    float v = gelu_exact(acc[t][j] + bv);
                    Hout[(size_t)(slotBase + m) * INNERD + n] = (_Float16)v;
                } else {
                    Yout[(size_t)(slotBase + m) * DIMD + n] = acc[t][j] + bv;
                }
            }
        }
    }
}

// ---------------------------------------------------------------------------
// 7) combine: out[t] = w0*Y[slot(e0)] + w1*Y[slot(e1)] + Y[shared slot]
// ---------------------------------------------------------------------------
__global__ __launch_bounds__(256) void combine_kernel(
        const float* __restrict__ Ybuf, const int4* __restrict__ route,
        const float2* __restrict__ routeW, const int* __restrict__ offs,
        float* __restrict__ out) {
    const int t = blockIdx.x;
    __shared__ int g0s, g1s, gss;
    __shared__ float w0s, w1s;
    if (threadIdx.x == 0) {
        int4 r = route[t]; float2 w = routeW[t];
        g0s = offs[r.x] + r.y;
        g1s = offs[r.z] + r.w;
        gss = offs[NEXP] + t;
        w0s = w.x; w1s = w.y;
    }
    __syncthreads();
    const float* y0 = Ybuf + (size_t)g0s * DIMD;
    const float* y1 = Ybuf + (size_t)g1s * DIMD;
    const float* ys = Ybuf + (size_t)gss * DIMD;
    for (int d = threadIdx.x; d < DIMD; d += 256)
        out[(size_t)t * DIMD + d] = w0s * y0[d] + w1s * y1[d] + ys[d];
}

// ---------------------------------------------------------------------------
// launch
// ---------------------------------------------------------------------------
extern "C" void kernel_launch(void* const* d_in, const int* in_sizes, int n_in,
                              void* d_out, int out_size, void* d_ws, size_t ws_size,
                              hipStream_t stream) {
    const float* x   = (const float*)d_in[0];   // [TOK, DIM]
    const float* Wg  = (const float*)d_in[1];   // [8, DIM]
    const float* W1  = (const float*)d_in[2];   // [8, DIM, INNER]
    const float* b1  = (const float*)d_in[3];   // [8, INNER]
    const float* W2  = (const float*)d_in[4];   // [8, INNER, DIM]
    const float* b2  = (const float*)d_in[5];   // [8, DIM]
    const float* sW1 = (const float*)d_in[6];   // [DIM, INNER]
    const float* sb1 = (const float*)d_in[7];   // [INNER]
    const float* sW2 = (const float*)d_in[8];   // [INNER, DIM]
    const float* sb2 = (const float*)d_in[9];   // [DIM]
    float* out = (float*)d_out;

    char* p = (char*)d_ws;
    auto alloc = [&](size_t bytes) -> void* {
        void* r = (void*)p;
        p += (bytes + 255) & ~(size_t)255;
        return r;
    };
    int*    counts  = (int*)alloc(9 * sizeof(int));
    int*    offs    = (int*)alloc(9 * sizeof(int));
    int*    idxList = (int*)alloc((size_t)NEXP * CAP * sizeof(int));
    int4*   route   = (int4*)alloc((size_t)TOK * sizeof(int4));
    float2* routeW  = (float2*)alloc((size_t)TOK * sizeof(float2));
    _Float16* Xh   = (_Float16*)alloc((size_t)TOK * DIMD * 2);
    _Float16* W1T  = (_Float16*)alloc((size_t)NEXP * INNERD * DIMD * 2);
    _Float16* W2T  = (_Float16*)alloc((size_t)NEXP * DIMD * INNERD * 2);
    _Float16* sW1T = (_Float16*)alloc((size_t)INNERD * DIMD * 2);
    _Float16* sW2T = (_Float16*)alloc((size_t)DIMD * INNERD * 2);
    _Float16* Hbuf = (_Float16*)alloc((size_t)(3 * TOK) * INNERD * 2);  // 2*TOK routed + TOK shared
    float*    Ybuf = (float*)alloc((size_t)(3 * TOK) * DIMD * 4);

    zero_counts_kernel<<<1, 32, 0, stream>>>(counts);
    gate_route_kernel<<<TOK, 256, 0, stream>>>(x, Wg, counts, idxList, route, routeW);
    offsets_kernel<<<1, 32, 0, stream>>>(counts, offs);

    cvt_f32_f16_kernel<<<(TOK * DIMD / 4 + 255) / 256, 256, 0, stream>>>(x, Xh, TOK * DIMD / 4);
    transpose_cvt_kernel<<<dim3(INNERD / 32, DIMD / 32, NEXP), dim3(32, 8), 0, stream>>>(W1, W1T, DIMD, INNERD);
    transpose_cvt_kernel<<<dim3(DIMD / 32, INNERD / 32, NEXP), dim3(32, 8), 0, stream>>>(W2, W2T, INNERD, DIMD);
    transpose_cvt_kernel<<<dim3(INNERD / 32, DIMD / 32, 1), dim3(32, 8), 0, stream>>>(sW1, sW1T, DIMD, INNERD);
    transpose_cvt_kernel<<<dim3(DIMD / 32, INNERD / 32, 1), dim3(32, 8), 0, stream>>>(sW2, sW2T, INNERD, DIMD);

    ffn_gemm_kernel<1><<<dim3(INNERD / BN, TOK / BM, NEXP + 1), 256, 0, stream>>>(
        Xh, W1T, sW1T, b1, sb1, counts, offs, idxList, Hbuf, nullptr);
    ffn_gemm_kernel<2><<<dim3(DIMD / BN, TOK / BM, NEXP + 1), 256, 0, stream>>>(
        Hbuf, W2T, sW2T, b2, sb2, counts, offs, nullptr, nullptr, Ybuf);
    combine_kernel<<<TOK, 256, 0, stream>>>(Ybuf, route, routeW, offs, out);
}
